// GraphAttentionLayer_33509334843571
// MI455X (gfx1250) — matrix operand order repaired and checked by reading, not compiled, not run on hardware
//
#include <hip/hip_runtime.h>
#include <hip/hip_bf16.h>

typedef __attribute__((ext_vector_type(16))) _Float16 v16h;
typedef __attribute__((ext_vector_type(8)))  float    v8f;

#define BN  4
#define NPT 2048
#define FIN 128
#define FO  64
#define NH  4
#define DD  256   // FO*NH

// --- WMMA fragment element maps (CDNA5 ISA 7.12.2, wave32) ---
// A (16x32 f16): lane holds row m=lane&15; half i -> K
__device__ __forceinline__ int kmapA(int lane, int i) {
    return (i & 7) + ((i >> 3) << 4) + ((lane >> 4) << 3);
}
// B (32x16 f16): lane holds col n=lane&15; half i -> K
__device__ __forceinline__ int kmapB(int lane, int i) {
    return (i & 7) + ((i >> 3) << 4) + ((lane >> 4) << 3);
}
// C/D (16x16 f32): vgpr r -> row m = r + 8*(lane>>4), col n = lane&15

__device__ __forceinline__ float elu1(float v) {
    return v > 0.f ? v : (__expf(v) - 1.f);
}

// ---------------------------------------------------------------------------
// Kernel 1: proj = x @ W_proj^T  (M=8192, K=128, N=256), WMMA f16, f32 accum.
// Writes proj in head-major f16 layout projh[b][h][n][f]  (d = f*4 + h).
// One wave per 16x16 output tile.
// ---------------------------------------------------------------------------
__global__ void __launch_bounds__(32)
gat_proj_gemm(const float* __restrict__ x, const float* __restrict__ Wp,
              _Float16* __restrict__ projh) {
    const int lane = threadIdx.x & 31;
    const int mt = blockIdx.x;   // 0..511 : 16-row tile of (bn*n)
    const int nt = blockIdx.y;   // 0..15  : 16-col tile of D
    const int m  = lane & 15;
    const int gm = mt * 16 + m;      // A row for this lane
    const int dn = nt * 16 + m;      // B col (= d index) for this lane

    v8f c = {};
    #pragma unroll
    for (int kk = 0; kk < FIN; kk += 32) {
        v16h a, b;
        #pragma unroll
        for (int i = 0; i < 16; ++i) {
            a[i] = (_Float16)x [ (size_t)gm * FIN + kk + kmapA(lane, i)];
            b[i] = (_Float16)Wp[ (size_t)dn * FIN + kk + kmapB(lane, i)];
        }
        c = __builtin_amdgcn_wmma_f32_16x16x32_f16(false, a, false, b,
                                                   (short)0, c, false, false);
    }

    const int dcol = nt * 16 + (lane & 15);
    const int hh = dcol & 3;      // head (fastest-varying in D split)
    const int ff = dcol >> 2;     // feature within head
    #pragma unroll
    for (int r = 0; r < 8; ++r) {
        const int row = mt * 16 + r + ((lane >> 4) << 3);
        const int bb  = row >> 11;        // / 2048
        const int nn  = row & (NPT - 1);
        projh[(((size_t)(bb * NH + hh) * NPT) + nn) * FO + ff] = (_Float16)c[r];
    }
}

// ---------------------------------------------------------------------------
// Kernel 2: s_src[b,h,n] = <proj[b,h,n,:], score_src[h,:]>, same for s_tgt.
// ---------------------------------------------------------------------------
__global__ void gat_scores(const _Float16* __restrict__ projh,
                           const float* __restrict__ wsrc,
                           const float* __restrict__ wtgt,
                           float* __restrict__ s_src,
                           float* __restrict__ s_tgt) {
    const int idx = blockIdx.x * blockDim.x + threadIdx.x;   // bh*NPT + n
    if (idx >= BN * NH * NPT) return;
    const int h = (idx / NPT) & (NH - 1);
    const _Float16* p = projh + (size_t)idx * FO;
    float a = 0.f, b = 0.f;
    #pragma unroll 8
    for (int f = 0; f < FO; ++f) {
        const float v = (float)p[f];
        a += v * wsrc[h * FO + f];
        b += v * wtgt[h * FO + f];
    }
    s_src[idx] = a;
    s_tgt[idx] = b;
}

// ---------------------------------------------------------------------------
// Kernel 3: flash attention. Block = (b, 16-row i-tile); 4 waves = 4 heads
// sharing each adj tile through LDS (adj read exactly once from HBM).
// Online softmax; P@V via 4x v_wmma_f32_16x16x32_f16 per 32-wide j chunk.
// Epilogue: elu(out/l + bias).
// ---------------------------------------------------------------------------
__global__ void __launch_bounds__(128)
gat_attention(const float* __restrict__ adj,
              const _Float16* __restrict__ projh,
              const float* __restrict__ s_src,
              const float* __restrict__ s_tgt,
              const float* __restrict__ bias,
              float* __restrict__ out) {
    __shared__ float adjT[16][33];     // padded vs bank conflicts
    __shared__ float stgtS[NH][32];

    const int tid  = threadIdx.x;
    const int lane = tid & 31;
    const int h    = tid >> 5;            // wave id == head
    const int it   = blockIdx.x;          // 0..127 i-tile
    const int b    = blockIdx.y;          // 0..3
    const int m    = lane & 15;
    const int bh   = b * NH + h;

    const float  ssrc_m    = s_src[(size_t)bh * NPT + it * 16 + m];
    const float* stgt_base = s_tgt + (size_t)bh * NPT;
    const float* adj_base  = adj + ((size_t)b * NPT + it * 16) * NPT;
    const _Float16* pv     = projh + (size_t)bh * NPT * FO;

    float mrun = -__builtin_inff();
    float lrun = 0.f;
    v8f acc0 = {}, acc1 = {}, acc2 = {}, acc3 = {};

    for (int j0 = 0; j0 < NPT; j0 += 32) {
        __syncthreads();   // protect LDS reuse from previous iteration
        #pragma unroll
        for (int q = 0; q < 4; ++q) {
            const int e = tid + q * 128;           // 512 floats, coalesced
            adjT[e >> 5][e & 31] =
                adj_base[(size_t)(e >> 5) * NPT + j0 + (e & 31)];
        }
        stgtS[h][lane] = stgt_base[j0 + lane];
        if (j0 + 32 < NPT)   // hint next adj tile toward caches
            __builtin_prefetch(&adj_base[(size_t)(tid >> 5) * NPT + j0 + 32], 0, 0);
        __syncthreads();

        // scores for this 16x32 tile (each lane: one row, 16 j's)
        float s[16];
        float tmax = -__builtin_inff();
        #pragma unroll
        for (int i = 0; i < 16; ++i) {
            const int k = kmapA(lane, i);
            float sv = ssrc_m + stgtS[h][k];
            sv = sv > 0.f ? sv : 0.2f * sv;        // leaky_relu(0.2)
            sv += adjT[m][k];
            s[i] = sv;
            tmax = fmaxf(tmax, sv);
        }
        tmax = fmaxf(tmax, __shfl_xor(tmax, 16, 32));   // join lane pair (row halves)
        const float newmax = fmaxf(mrun, tmax);
        const float corr   = __expf(mrun - newmax);
        mrun = newmax;

        v16h pa;
        float psum = 0.f;
        #pragma unroll
        for (int i = 0; i < 16; ++i) {
            const float e = __expf(s[i] - newmax);
            pa[i] = (_Float16)e;
            psum += e;
        }
        psum += __shfl_xor(psum, 16, 32);
        lrun = lrun * corr + psum;

        // rescale accumulators: row of C element r is  r + 8*(lane>>4)
        float corr_r[8];
        #pragma unroll
        for (int r = 0; r < 8; ++r)
            corr_r[r] = __shfl(corr, r + ((lane >> 4) << 3), 32);
        #pragma unroll
        for (int r = 0; r < 8; ++r) {
            acc0[r] *= corr_r[r]; acc1[r] *= corr_r[r];
            acc2[r] *= corr_r[r]; acc3[r] *= corr_r[r];
        }

        // V fragments (B operand): proj[b,h, j0+K, f-tile*16 + (lane&15)]
        v16h b0, b1, b2, b3;
        const int f0 = lane & 15;
        #pragma unroll
        for (int i = 0; i < 16; ++i) {
            const size_t jrow = (size_t)(j0 + kmapB(lane, i)) * FO;
            b0[i] = pv[jrow +  0 + f0];
            b1[i] = pv[jrow + 16 + f0];
            b2[i] = pv[jrow + 32 + f0];
            b3[i] = pv[jrow + 48 + f0];
        }
        acc0 = __builtin_amdgcn_wmma_f32_16x16x32_f16(false, pa, false, b0, (short)0, acc0, false, false);
        acc1 = __builtin_amdgcn_wmma_f32_16x16x32_f16(false, pa, false, b1, (short)0, acc1, false, false);
        acc2 = __builtin_amdgcn_wmma_f32_16x16x32_f16(false, pa, false, b2, (short)0, acc2, false, false);
        acc3 = __builtin_amdgcn_wmma_f32_16x16x32_f16(false, pa, false, b3, (short)0, acc3, false, false);
    }

    // epilogue: divide by softmax denom, add bias, ELU, write (b, n, h*64+f)
    const int f0 = lane & 15;
    const float bia0 = bias[h * FO +  0 + f0];
    const float bia1 = bias[h * FO + 16 + f0];
    const float bia2 = bias[h * FO + 32 + f0];
    const float bia3 = bias[h * FO + 48 + f0];
    #pragma unroll
    for (int r = 0; r < 8; ++r) {
        const float l_r = __shfl(lrun, r + ((lane >> 4) << 3), 32);
        const float inv = 1.f / l_r;
        const int row = it * 16 + r + ((lane >> 4) << 3);
        float* orow = out + ((size_t)b * NPT + row) * DD + h * FO;
        orow[ 0 + f0] = elu1(acc0[r] * inv + bia0);
        orow[16 + f0] = elu1(acc1[r] * inv + bia1);
        orow[32 + f0] = elu1(acc2[r] * inv + bia2);
        orow[48 + f0] = elu1(acc3[r] * inv + bia3);
    }
}

// ---------------------------------------------------------------------------
extern "C" void kernel_launch(void* const* d_in, const int* in_sizes, int n_in,
                              void* d_out, int out_size, void* d_ws, size_t ws_size,
                              hipStream_t stream) {
    (void)in_sizes; (void)n_in; (void)out_size; (void)ws_size;
    const float* x    = (const float*)d_in[0];
    const float* adj  = (const float*)d_in[1];
    const float* Wp   = (const float*)d_in[2];
    const float* wsrc = (const float*)d_in[3];
    const float* wtgt = (const float*)d_in[4];
    const float* bias = (const float*)d_in[5];
    // d_in[6..12] = GRU gate weights: computed-then-discarded in the
    // reference, so they never touch the output — skipped entirely.

    // workspace layout
    _Float16* projh = (_Float16*)d_ws;                              // 4 MB
    float* s_src = (float*)((char*)d_ws +
                   (size_t)BN * NH * NPT * FO * sizeof(_Float16));  // 128 KB
    float* s_tgt = s_src + (size_t)BN * NH * NPT;                   // 128 KB
    float* out = (float*)d_out;

    gat_proj_gemm<<<dim3(512, 16), 32, 0, stream>>>(x, Wp, projh);
    gat_scores<<<(BN * NH * NPT + 255) / 256, 256, 0, stream>>>(projh, wsrc, wtgt,
                                                                s_src, s_tgt);
    gat_attention<<<dim3(NPT / 16, BN), 128, 0, stream>>>(adj, projh, s_src,
                                                          s_tgt, bias, out);
}
